// nmr_mpnn_BASELINE_84026740179727
// MI455X (gfx1250) — matrix operand order repaired and checked by reading, not compiled, not run on hardware
//
#include <hip/hip_runtime.h>
#include <hip/hip_bf16.h>

// ---------------------------------------------------------------------------
// MPNN forward for MI455X (gfx1250), wave32 + WMMA bf16.
// Dense layers: v_wmma_f32_16x16x32_bf16 (bf16 in, fp32 accum).
// LDS tiles staged in WMMA fragment order (fragment load = 2x ds_load_b128),
// double-buffered LDS with register prefetch, guard-free hot path.
// ---------------------------------------------------------------------------

#define Nn   4096
#define Gg   128
#define NPG  32
#define Ee   8192
#define NI   64
#define EI   16
#define NF   64
#define NH   256
#define PH   512
#define Tt   5

typedef __attribute__((ext_vector_type(16))) __bf16    v16bf;
typedef __attribute__((ext_vector_type(8)))  float     v8f;
typedef __attribute__((ext_vector_type(8)))  unsigned  v8u;

union Frag16 { v16bf v; v8u u8; unsigned u[8]; };

__device__ __forceinline__ unsigned short f2bf(float f) {
    union { float f; unsigned u; } x; x.f = f;
    unsigned r = x.u + 0x7FFFu + ((x.u >> 16) & 1u);   // RNE
    return (unsigned short)(r >> 16);
}
__device__ __forceinline__ float bf2f(unsigned short h) {
    union { unsigned u; float f; } x; x.u = ((unsigned)h) << 16;
    return x.f;
}

// ---------------------------------------------------------------------------
// Tiled GEMM: C[M x Nc] = act( A[M x K](bf16) * B[K x Nc](bf16) + bias )
// Block = 128 threads = 4 waves; block tile 128x64; wave tile 32x64
// (2 A-fragments x 4 B-fragments = 8 WMMAs per k-step).
// REQUIRES: M % 128 == 0, Nc % 64 == 0, K even (all launches satisfy this).
// act: 0=none 1=relu 2=tanh 3=prelu(alpha=prelu[prelu_idx])
// ---------------------------------------------------------------------------

struct StageRegs {
    unsigned a[16];
    unsigned b0[4], b1[4];
};

// Guard-free global->register staging (k-tile fully inside K).
__device__ __forceinline__ void stage_full(
    StageRegs& r, const unsigned short* __restrict__ A, int lda,
    const unsigned short* __restrict__ B, int ldb,
    int m0, int n0, int k0, int tid)
{
    #pragma unroll
    for (int i = 0; i < 16; ++i) {
        int d  = i * 128 + tid;       // (m, j2)
        int m  = d >> 4;
        int j2 = d & 15;
        r.a[i] = *(const unsigned*)(A + (size_t)(m0 + m) * lda + (k0 + 2 * j2));
    }
    #pragma unroll
    for (int i = 0; i < 4; ++i) {
        int p  = i * 128 + tid;       // (j2, col-pair)
        int j2 = p >> 5;
        int np = p & 31;
        int gk = k0 + 2 * j2;
        int gn = n0 + np * 2;
        r.b0[i] = *(const unsigned*)(B + (size_t)gk * ldb + gn);
        r.b1[i] = *(const unsigned*)(B + (size_t)(gk + 1) * ldb + gn);
    }
}

// Branch-free tail staging: clamp address into range, select 0 for OOB.
// (No exec-mask divergence: unconditional load + v_cndmask.)
__device__ __forceinline__ void stage_tail(
    StageRegs& r, const unsigned short* __restrict__ A, int lda,
    const unsigned short* __restrict__ B, int ldb,
    int m0, int n0, int k0, int K, int tid)
{
    #pragma unroll
    for (int i = 0; i < 16; ++i) {
        int d  = i * 128 + tid;
        int m  = d >> 4;
        int j2 = d & 15;
        int gk  = k0 + 2 * j2;
        int gkc = (gk < K) ? gk : k0;              // k0 < K always
        unsigned v = *(const unsigned*)(A + (size_t)(m0 + m) * lda + gkc);
        r.a[i] = (gk < K) ? v : 0u;
    }
    #pragma unroll
    for (int i = 0; i < 4; ++i) {
        int p  = i * 128 + tid;
        int j2 = p >> 5;
        int np = p & 31;
        int gk  = k0 + 2 * j2;
        int gkc = (gk < K) ? gk : k0;
        int gn  = n0 + np * 2;
        unsigned v0 = *(const unsigned*)(B + (size_t)gkc * ldb + gn);
        unsigned v1 = *(const unsigned*)(B + (size_t)(gkc + 1) * ldb + gn);
        r.b0[i] = (gk < K) ? v0 : 0u;
        r.b1[i] = (gk < K) ? v1 : 0u;
    }
}

// Register->LDS store in fragment order.
__device__ __forceinline__ void stage_store(
    const StageRegs& r, unsigned* sAf, unsigned* sBf, int tid)
{
    #pragma unroll
    for (int i = 0; i < 16; ++i) {
        int d  = i * 128 + tid;
        int m  = d >> 4;
        int j2 = d & 15;
        int w  = m >> 4;
        int ln = (((j2 >> 2) & 1) << 4) | (m & 15);
        int j  = (j2 & 3) | ((j2 >> 3) << 2);
        sAf[((w * 32 + ln) << 3) + j] = r.a[i];
    }
    #pragma unroll
    for (int i = 0; i < 4; ++i) {
        int p  = i * 128 + tid;
        int j2 = p >> 5;
        int np = p & 31;
        unsigned v0 = (r.b0[i] & 0xffffu) | (r.b1[i] << 16);
        unsigned v1 = (r.b0[i] >> 16) | (r.b1[i] & 0xffff0000u);
        int n  = np * 2;
        int ct = n >> 4;
        int ln = (((j2 >> 2) & 1) << 4) | (n & 15);
        int base = ((ct * 32 + ln) << 3) + ((j2 & 3) | ((j2 >> 3) << 2));
        sBf[base]     = v0;
        sBf[base + 8] = v1;            // col n+1 -> lane+1 (same subtile)
    }
}

__device__ __forceinline__ void compute_tile(
    const unsigned* sAf, const unsigned* sBf, int wave, int lane,
    v8f acc0[4], v8f acc1[4])
{
    Frag16 a0, a1;
    a0.u8 = *((const v8u*)(sAf + (((wave * 2 + 0) * 32 + lane) << 3)));
    a1.u8 = *((const v8u*)(sAf + (((wave * 2 + 1) * 32 + lane) << 3)));
    #pragma unroll
    for (int ct = 0; ct < 4; ++ct) {
        Frag16 bfr;
        bfr.u8 = *((const v8u*)(sBf + ((ct * 32 + lane) << 3)));
        acc0[ct] = __builtin_amdgcn_wmma_f32_16x16x32_bf16(
            false, a0.v, false, bfr.v, (short)0, acc0[ct], false, false);
        acc1[ct] = __builtin_amdgcn_wmma_f32_16x16x32_bf16(
            false, a1.v, false, bfr.v, (short)0, acc1[ct], false, false);
    }
}

__global__ __launch_bounds__(128) void gemm_bf16_wmma(
    const unsigned short* __restrict__ A, int lda,
    const unsigned short* __restrict__ B, int ldb,
    const float* __restrict__ bias,
    float* __restrict__ Cf, int ldcf,
    unsigned short* __restrict__ Cb, int ldcb,
    int M, int Nc, int K,
    int act, const float* __restrict__ prelu, int prelu_idx)
{
    // Fragment-order LDS, double buffered: slot = (subtile*32 + lane), 8 dwords.
    __shared__ unsigned sAf[2][8 * 32 * 8];   // 2 x 8 KB
    __shared__ unsigned sBf[2][4 * 32 * 8];   // 2 x 4 KB

    const int tid  = threadIdx.x;
    const int wave = tid >> 5;
    const int lane = tid & 31;
    const int m0   = blockIdx.x * 128;
    const int n0   = blockIdx.y * 64;
    const int lane15 = lane & 15;
    const int laneHi = lane >> 4;

    v8f acc0[4], acc1[4];
    #pragma unroll
    for (int j = 0; j < 4; ++j) {
        acc0[j] = (v8f){0.f,0.f,0.f,0.f,0.f,0.f,0.f,0.f};
        acc1[j] = (v8f){0.f,0.f,0.f,0.f,0.f,0.f,0.f,0.f};
    }

    const int kTiles = (K + 31) >> 5;
    StageRegs r;

    // prologue: stage tile 0
    if (32 <= K) stage_full(r, A, lda, B, ldb, m0, n0, 0, tid);
    else         stage_tail(r, A, lda, B, ldb, m0, n0, 0, K, tid);
    stage_store(r, sAf[0], sBf[0], tid);
    __syncthreads();

    for (int t = 0; t < kTiles; ++t) {
        int cur = t & 1;
        if (t + 1 < kTiles) {
            int k0 = (t + 1) << 5;
            if (k0 + 32 <= K) stage_full(r, A, lda, B, ldb, m0, n0, k0, tid);
            else              stage_tail(r, A, lda, B, ldb, m0, n0, k0, K, tid);
        }
        compute_tile(sAf[cur], sBf[cur], wave, lane, acc0, acc1);
        if (t + 1 < kTiles) {
            stage_store(r, sAf[cur ^ 1], sBf[cur ^ 1], tid);
            __syncthreads();
        }
    }

    // ---- epilogue: bias + activation, fp32 and/or bf16 stores -----------
    float alpha = 0.0f;
    if (act == 3) alpha = prelu[prelu_idx];
    #pragma unroll
    for (int half = 0; half < 2; ++half) {
        int rbase = m0 + wave * 32 + half * 16 + (laneHi ? 8 : 0);
        #pragma unroll
        for (int ct = 0; ct < 4; ++ct) {
            int col = n0 + (ct << 4) + lane15;
            float bv = bias ? bias[col] : 0.0f;
            #pragma unroll
            for (int v = 0; v < 8; ++v) {
                int row = rbase + v;
                float x = (half ? acc1[ct][v] : acc0[ct][v]) + bv;
                if      (act == 1) x = fmaxf(x, 0.0f);
                else if (act == 2) x = tanhf(x);
                else if (act == 3) x = (x > 0.0f) ? x : alpha * x;
                if (Cf) Cf[(size_t)row * ldcf + col] = x;
                if (Cb) Cb[(size_t)row * ldcb + col] = f2bf(x);
            }
        }
    }
}

// ---------------------------------------------------------------------------
__global__ void cvt_f32_bf16_k(unsigned short* __restrict__ dst,
                               const float* __restrict__ src, int n) {
    int i = blockIdx.x * blockDim.x + threadIdx.x;
    if (i < n) dst[i] = f2bf(src[i]);
}

__global__ void zero_f32_k(float* __restrict__ p, int n) {
    int i = blockIdx.x * blockDim.x + threadIdx.x;
    if (i < n) p[i] = 0.0f;
}

// copy bf16 matrix rows with differing leading dims (places h into a1 slot 0)
__global__ void copy_bf_rows_k(unsigned short* __restrict__ dst, int ldd,
                               const unsigned short* __restrict__ src, int lds_,
                               int rows, int cols) {
    int i = blockIdx.x * blockDim.x + threadIdx.x;
    if (i >= rows * cols) return;
    int r = i / cols, c = i - r * cols;
    dst[(size_t)r * ldd + c] = src[(size_t)r * lds_ + c];
}

// per-edge m = h[src] (1x64) * eW[e] (64x64), scatter-add into msg[dst]
__global__ __launch_bounds__(256) void edge_message_k(
    const unsigned short* __restrict__ h_bf,
    const unsigned short* __restrict__ eW,
    const int* __restrict__ src, const int* __restrict__ dst,
    float* __restrict__ msg, int nE)
{
    __shared__ float sh[4][NF];
    int t  = threadIdx.x;
    int el = t >> 6;              // edge-in-block 0..3
    int o  = t & 63;              // output column
    int e  = blockIdx.x * 4 + el;
    bool valid = (e < nE);
    int s = 0, d = 0;
    if (valid) { s = src[e]; d = dst[e]; }
    sh[el][o] = valid ? bf2f(h_bf[(size_t)s * NF + o]) : 0.0f;
    __syncthreads();
    if (!valid) return;
    const unsigned short* W = eW + (size_t)e * (NF * NF);
    float acc = 0.0f;
    #pragma unroll 8
    for (int i = 0; i < NF; ++i)
        acc += sh[el][i] * bf2f(W[i * NF + o]);
    atomicAdd(&msg[(size_t)d * NF + o], acc);
}

// msg_bf = bf16(msg + conv_bias)
__global__ void msg_prep_k(unsigned short* __restrict__ msg_bf,
                           const float* __restrict__ msg,
                           const float* __restrict__ conv_bias, int n) {
    int i = blockIdx.x * blockDim.x + threadIdx.x;
    if (i < n) msg_bf[i] = f2bf(msg[i] + conv_bias[i & (NF - 1)]);
}

// fused GRU gates; writes h(fp32), h(bf16), and a1 slot in rn_in
__global__ void gru_k(const float* __restrict__ gi, const float* __restrict__ gh,
                      float* __restrict__ h, unsigned short* __restrict__ h_bf,
                      unsigned short* __restrict__ rn_in, int slot, int n) {
    int i = blockIdx.x * blockDim.x + threadIdx.x;
    if (i >= n) return;
    int node = i >> 6, o = i & 63;
    const float* gir = gi + (size_t)node * (3 * NF);
    const float* ghr = gh + (size_t)node * (3 * NF);
    float ir = gir[o], iz = gir[NF + o], inn = gir[2 * NF + o];
    float hr = ghr[o], hz = ghr[NF + o], hn  = ghr[2 * NF + o];
    float r = 1.0f / (1.0f + expf(-(ir + hr)));
    float z = 1.0f / (1.0f + expf(-(iz + hz)));
    float nn = tanhf(inn + r * hn);
    float hv = (1.0f - z) * nn + z * h[i];
    h[i] = hv;
    unsigned short b = f2bf(hv);
    h_bf[i] = b;
    rn_in[(size_t)node * (NF * (1 + Tt) + PH) + slot * NF + o] = b;
}

// a2 = concat(node_attr_bf[64], h_bf[64])  -> [N,128] bf16
__global__ void concat_a2_k(unsigned short* __restrict__ a2,
                            const unsigned short* __restrict__ na,
                            const unsigned short* __restrict__ hbf, int n) {
    int i = blockIdx.x * blockDim.x + threadIdx.x;
    if (i >= n) return;
    int node = i >> 7, c = i & 127;
    a2[i] = (c < NF) ? na[(size_t)node * NI + c]
                     : hbf[(size_t)node * NF + (c - NF)];
}

// graph pooling: gfeat[g] = sum over 32 consecutive nodes (graph_ids = repeat)
__global__ __launch_bounds__(512) void pool_k(float* __restrict__ gfeat,
                                              const float* __restrict__ gproj) {
    int g = blockIdx.x, c = threadIdx.x;
    float s = 0.0f;
    #pragma unroll 8
    for (int i = 0; i < NPG; ++i)
        s += gproj[(size_t)(g * NPG + i) * PH + c];
    gfeat[(size_t)g * PH + c] = s;
}

// broadcast pooled feature back per node into rn_in columns [384, 896)
__global__ void gnode_k(unsigned short* __restrict__ rn_in,
                        const float* __restrict__ gfeat,
                        const int* __restrict__ gids, int n) {
    int i = blockIdx.x * blockDim.x + threadIdx.x;
    if (i >= n) return;
    int node = i >> 9, c = i & 511;
    rn_in[(size_t)node * (NF * (1 + Tt) + PH) + NF * (1 + Tt) + c] =
        f2bf(gfeat[(size_t)gids[node] * PH + c]);
}

// final 512 -> 1 layer: one wave per node, lane-strided dot + shfl reduce
__global__ __launch_bounds__(256) void final_dot_k(
    float* __restrict__ out, const unsigned short* __restrict__ x,
    const unsigned short* __restrict__ w, const float* __restrict__ b,
    const unsigned char* __restrict__ masks, int nNodes) {
    int wave = threadIdx.x >> 5, lane = threadIdx.x & 31;
    int node = blockIdx.x * 8 + wave;
    if (node >= nNodes) return;
    float acc = 0.0f;
    #pragma unroll
    for (int j = 0; j < 16; ++j) {
        int c = lane * 16 + j;
        acc += bf2f(x[(size_t)node * PH + c]) * bf2f(w[c]);
    }
    #pragma unroll
    for (int off = 16; off > 0; off >>= 1)
        acc += __shfl_down(acc, off, 32);
    if (lane == 0) {
        float v = acc + b[0];
        out[node] = masks[node] ? v : 0.0f;
    }
}

// ---------------------------------------------------------------------------
extern "C" void kernel_launch(void* const* d_in, const int* in_sizes, int n_in,
                              void* d_out, int out_size, void* d_ws, size_t ws_size,
                              hipStream_t stream) {
    // ---- inputs (setup_inputs dict order, nested lists flattened) -------
    const float* node_attr = (const float*)d_in[0];
    const float* edge_attr = (const float*)d_in[1];
    const int*   src       = (const int*)d_in[2];
    const int*   dst       = (const int*)d_in[3];
    const int*   gids      = (const int*)d_in[4];
    const unsigned char* masks = (const unsigned char*)d_in[5];
    const float* pW[4] = {(const float*)d_in[6],  (const float*)d_in[8],
                          (const float*)d_in[10], (const float*)d_in[12]};
    const float* pB[4] = {(const float*)d_in[7],  (const float*)d_in[9],
                          (const float*)d_in[11], (const float*)d_in[13]};
    const float* eWp[4] = {(const float*)d_in[14], (const float*)d_in[16],
                           (const float*)d_in[18], (const float*)d_in[20]};
    const float* eBp[4] = {(const float*)d_in[15], (const float*)d_in[17],
                           (const float*)d_in[19], (const float*)d_in[21]};
    const float* conv_bias = (const float*)d_in[22];
    const float* gWih = (const float*)d_in[23];
    const float* gWhh = (const float*)d_in[24];
    const float* gBih = (const float*)d_in[25];
    const float* gBhh = (const float*)d_in[26];
    const float* rgW[4] = {(const float*)d_in[27], (const float*)d_in[29],
                           (const float*)d_in[31], (const float*)d_in[33]};
    const float* rgB[4] = {(const float*)d_in[28], (const float*)d_in[30],
                           (const float*)d_in[32], (const float*)d_in[34]};
    const float* rnW[4] = {(const float*)d_in[35], (const float*)d_in[37],
                           (const float*)d_in[39], (const float*)d_in[41]};
    const float* rnB[4] = {(const float*)d_in[36], (const float*)d_in[38],
                           (const float*)d_in[40], (const float*)d_in[42]};
    const float* rn_prelu = (const float*)d_in[43];
    float* out = (float*)d_out;

    // ---- workspace bump allocator ---------------------------------------
    char* ws = (char*)d_ws;
    size_t off = 0;
    auto alloc = [&](size_t bytes) -> void* {
        void* p = ws + off;
        off = (off + bytes + 255) & ~(size_t)255;
        return p;
    };
    auto aU = [&](size_t n) { return (unsigned short*)alloc(n * 2); };
    auto aF = [&](size_t n) { return (float*)alloc(n * 4); };

    const int RNW = NF * (1 + Tt) + PH;   // 896

    unsigned short* na_bf  = aU((size_t)Nn * NI);
    unsigned short* ea_bf  = aU((size_t)Ee * EI);
    unsigned short* pWb[4], *eWb[4], *rgWb[4], *rnWb[4];
    const int pWn[4]  = {NI * NH, NH * NH, NH * NH, NH * NF};
    const int eWn[4]  = {EI * NH, NH * NH, NH * NH, NH * NF * NF};
    const int rgWn[4] = {(NF + NI) * PH, PH * PH, PH * PH, PH * PH};
    const int rnWn[4] = {RNW * PH, PH * PH, PH * PH, PH * 1};
    for (int i = 0; i < 4; ++i) pWb[i]  = aU(pWn[i]);
    for (int i = 0; i < 4; ++i) eWb[i]  = aU(eWn[i]);
    for (int i = 0; i < 4; ++i) rgWb[i] = aU(rgWn[i]);
    for (int i = 0; i < 4; ++i) rnWb[i] = aU(rnWn[i]);
    unsigned short* gWihb = aU(NF * 3 * NF);
    unsigned short* gWhhb = aU(NF * 3 * NF);

    unsigned short* etmp0 = aU((size_t)Ee * NH);
    unsigned short* etmp1 = aU((size_t)Ee * NH);
    unsigned short* eWmat = aU((size_t)Ee * NF * NF);     // 64 MB bf16
    unsigned short* tmp0  = aU((size_t)Nn * PH);
    unsigned short* tmp1  = aU((size_t)Nn * PH);
    unsigned short* h_bf  = aU((size_t)Nn * NF);
    unsigned short* msgbf = aU((size_t)Nn * NF);
    unsigned short* a2_bf = aU((size_t)Nn * (NF + NI));
    unsigned short* rn_in = aU((size_t)Nn * RNW);

    float* h_f   = aF((size_t)Nn * NF);
    float* msg_f = aF((size_t)Nn * NF);
    float* gi_f  = aF((size_t)Nn * 3 * NF);
    float* gh_f  = aF((size_t)Nn * 3 * NF);
    float* gproj = aF((size_t)Nn * PH);
    float* gfeat = aF((size_t)Gg * PH);

    // ---- helpers ---------------------------------------------------------
    auto cvt = [&](unsigned short* dstp, const float* srcp, int n) {
        cvt_f32_bf16_k<<<(n + 255) / 256, 256, 0, stream>>>(dstp, srcp, n);
    };
    auto gemm = [&](const unsigned short* A, int lda, const unsigned short* B, int ldb,
                    const float* bias, float* Cf, int ldcf,
                    unsigned short* Cb, int ldcb,
                    int M, int Ncc, int K, int act,
                    const float* pr = nullptr, int pi = 0) {
        dim3 grid((M + 127) / 128, (Ncc + 63) / 64);
        gemm_bf16_wmma<<<grid, 128, 0, stream>>>(A, lda, B, ldb, bias,
                                                 Cf, ldcf, Cb, ldcb,
                                                 M, Ncc, K, act, pr, pi);
    };

    // ---- convert inputs + weights to bf16 --------------------------------
    cvt(na_bf, node_attr, Nn * NI);
    cvt(ea_bf, edge_attr, Ee * EI);
    for (int i = 0; i < 4; ++i) cvt(pWb[i],  pW[i],  pWn[i]);
    for (int i = 0; i < 4; ++i) cvt(eWb[i],  eWp[i], eWn[i]);
    for (int i = 0; i < 4; ++i) cvt(rgWb[i], rgW[i], rgWn[i]);
    for (int i = 0; i < 4; ++i) cvt(rnWb[i], rnW[i], rnWn[i]);
    cvt(gWihb, gWih, NF * 3 * NF);
    cvt(gWhhb, gWhh, NF * 3 * NF);

    // ---- node projection MLP: 64->256->256->256->64 (relu^3, tanh) ------
    gemm(na_bf, NI, pWb[0], NH, pB[0], nullptr, 0, tmp0, NH, Nn, NH, NI, 1);
    gemm(tmp0,  NH, pWb[1], NH, pB[1], nullptr, 0, tmp1, NH, Nn, NH, NH, 1);
    gemm(tmp1,  NH, pWb[2], NH, pB[2], nullptr, 0, tmp0, NH, Nn, NH, NH, 1);
    gemm(tmp0,  NH, pWb[3], NF, pB[3], h_f, NF, h_bf, NF, Nn, NF, NH, 2);
    // a1 slot 0 = initial h
    copy_bf_rows_k<<<(Nn * NF + 255) / 256, 256, 0, stream>>>(
        rn_in, RNW, h_bf, NF, Nn, NF);

    // ---- edge MLP: 16->256->256->256->4096 (relu everywhere) ------------
    gemm(ea_bf, EI, eWb[0], NH, eBp[0], nullptr, 0, etmp0, NH, Ee, NH, EI, 1);
    gemm(etmp0, NH, eWb[1], NH, eBp[1], nullptr, 0, etmp1, NH, Ee, NH, NH, 1);
    gemm(etmp1, NH, eWb[2], NH, eBp[2], nullptr, 0, etmp0, NH, Ee, NH, NH, 1);
    gemm(etmp0, NH, eWb[3], NF * NF, eBp[3], nullptr, 0, eWmat, NF * NF,
         Ee, NF * NF, NH, 1);                       // 8192x4096x256 — the big one

    // ---- 5 message-passing + GRU steps ----------------------------------
    for (int t = 0; t < Tt; ++t) {
        zero_f32_k<<<(Nn * NF + 255) / 256, 256, 0, stream>>>(msg_f, Nn * NF);
        edge_message_k<<<Ee / 4, 256, 0, stream>>>(h_bf, eWmat, src, dst, msg_f, Ee);
        msg_prep_k<<<(Nn * NF + 255) / 256, 256, 0, stream>>>(
            msgbf, msg_f, conv_bias, Nn * NF);
        gemm(msgbf, NF, gWihb, 3 * NF, gBih, gi_f, 3 * NF, nullptr, 0,
             Nn, 3 * NF, NF, 0);
        gemm(h_bf,  NF, gWhhb, 3 * NF, gBhh, gh_f, 3 * NF, nullptr, 0,
             Nn, 3 * NF, NF, 0);
        gru_k<<<(Nn * NF + 255) / 256, 256, 0, stream>>>(
            gi_f, gh_f, h_f, h_bf, rn_in, t + 1, Nn * NF);
    }

    // ---- readout: graph projection + pooling ----------------------------
    concat_a2_k<<<(Nn * (NF + NI) + 255) / 256, 256, 0, stream>>>(
        a2_bf, na_bf, h_bf, Nn * (NF + NI));
    gemm(a2_bf, NF + NI, rgWb[0], PH, rgB[0], nullptr, 0, tmp0, PH,
         Nn, PH, NF + NI, 1);
    gemm(tmp0, PH, rgWb[1], PH, rgB[1], nullptr, 0, tmp1, PH, Nn, PH, PH, 1);
    gemm(tmp1, PH, rgWb[2], PH, rgB[2], nullptr, 0, tmp0, PH, Nn, PH, PH, 1);
    gemm(tmp0, PH, rgWb[3], PH, rgB[3], gproj, PH, nullptr, 0, Nn, PH, PH, 1);
    pool_k<<<Gg, PH, 0, stream>>>(gfeat, gproj);
    gnode_k<<<(Nn * PH + 255) / 256, 256, 0, stream>>>(rn_in, gfeat, gids, Nn * PH);

    // ---- regression head: 896->512->512->512->1 (PReLU x3) --------------
    gemm(rn_in, RNW, rnWb[0], PH, rnB[0], nullptr, 0, tmp0, PH,
         Nn, PH, RNW, 3, rn_prelu, 0);
    gemm(tmp0, PH, rnWb[1], PH, rnB[1], nullptr, 0, tmp1, PH,
         Nn, PH, PH, 3, rn_prelu, 1);
    gemm(tmp1, PH, rnWb[2], PH, rnB[2], nullptr, 0, tmp0, PH,
         Nn, PH, PH, 3, rn_prelu, 2);
    final_dot_k<<<Nn / 8, 256, 0, stream>>>(out, tmp0, rnWb[3], rnB[3], masks, Nn);

    (void)in_sizes; (void)n_in; (void)out_size; (void)ws_size;
}
